// CustomAttention_15006615733244
// MI455X (gfx1250) — matrix operand order repaired and checked by reading, not compiled, not run on hardware
//
#include <hip/hip_runtime.h>
#include <cstdint>

// ---------------------------------------------------------------------------
// MI455X (gfx1250) multi-head attention forward, bf16 WMMA with fp32 accumulate
// + TDM double-buffered flash attention + async-to-LDS GEMM staging
// ---------------------------------------------------------------------------
typedef __bf16 bf16;
typedef __attribute__((ext_vector_type(8)))  __bf16 v8bf;
typedef __attribute__((ext_vector_type(16))) __bf16 v16bf;
typedef __attribute__((ext_vector_type(8)))  float  v8f;
typedef __attribute__((ext_vector_type(4)))  float  v4f;
typedef unsigned int u32;
typedef __attribute__((ext_vector_type(4))) u32 u32x4;
typedef __attribute__((ext_vector_type(8))) u32 u32x8;

#define EMBED 1024
#define HEADS 16
#define HDIM 64
#define SEQ 2048
#define BATCH 2

__device__ __forceinline__ v16bf combine16(v8bf lo, v8bf hi) {
    v16bf r;
#pragma unroll
    for (int i = 0; i < 8; ++i) { r[i] = lo[i]; r[i + 8] = hi[i]; }
    return r;
}

// LDS byte address of a flat pointer into __shared__ (LDS aperture keeps the
// workgroup LDS offset in addr[31:0]).
__device__ __forceinline__ u32 lds_addr32(const void* p) {
    return (u32)(uintptr_t)p;
}

// Per-lane async global->LDS 16-byte copy (ASYNCcnt path).
__device__ __forceinline__ void async_load_b128(void* lds, const void* g) {
    asm volatile("global_load_async_to_lds_b128 %0, %1, off"
                 :: "v"(lds_addr32(lds)), "v"((unsigned long long)(uintptr_t)g)
                 : "memory");
}
__device__ __forceinline__ void wait_async0() {
    asm volatile("s_wait_asynccnt 0x0" ::: "memory");
}
__device__ __forceinline__ void wait_tensor0() {
    asm volatile("s_wait_tensorcnt 0x0" ::: "memory");
}

// One-instruction XOR-lane exchange (DS_SWIZZLE_B32 group-of-32 mode).
template <int XORMASK>
__device__ __forceinline__ float swz_xor(float v) {
    int i = __builtin_bit_cast(int, v);
    i = __builtin_amdgcn_ds_swizzle(i, (XORMASK << 10) | 0x1F);
    return __builtin_bit_cast(float, i);
}

// TDM: load a 64x64 bf16 tile (global row stride `row_stride` elements) into
// LDS with 32 B padding after each 128 B row -> LDS rows of 80 bf16.
// D# per ISA 8.3/8.4: pad_interval code 4 (32 dwords), pad_amount code 7 (8 dw).
// Arguments must be wave-uniform; issued once per calling wave (TENSORcnt).
__device__ __forceinline__ void tdm_load_tile64(u32 lds_byte_addr, const void* gptr,
                                                u32 row_stride) {
    unsigned long long ga = (unsigned long long)(uintptr_t)gptr;
    u32x4 g0;
    g0[0] = 1u;                                            // count=1, user mode
    g0[1] = lds_byte_addr;                                 // lds_addr
    g0[2] = (u32)ga;                                       // global_addr[31:0]
    g0[3] = (u32)((ga >> 32) & 0x01FFFFFFu) | (2u << 30);  // ga[56:32] | type=2
    u32x8 g1;
    g1[0] = (1u << 16) | (1u << 20) | (4u << 22) | (7u << 25);
    g1[1] = 64u << 16;                                     // tensor_dim0 = 64
    g1[2] = 64u << 16;                                     // tensor_dim1 = 64
    g1[3] = 64u << 16;                                     // tile_dim0 = 64
    g1[4] = 64u;                                           // tile_dim1 = 64
    g1[5] = row_stride;                                    // tensor_dim0_stride
    g1[6] = 0u;
    g1[7] = 0u;
    asm volatile("tensor_load_to_lds %0, %1" :: "s"(g0), "s"(g1) : "memory");
}

// ---------------------------------------------------------------------------
// Generic bf16-WMMA GEMM: C = A(MxK) * B(KxN) + bias
// MODE 0: write fp32 C
// MODE 1: scatter QKV -> q,k: [B,H,S,D]; v: [B,H,D,S] (transposed); Q pre-scaled
//         (which third of the columns this block feeds is block-uniform)
// ---------------------------------------------------------------------------
template <typename AT, int MODE>
__global__ __launch_bounds__(256) void gemm_bf16_wmma(
    const AT* __restrict__ A, const float* __restrict__ Bw,
    const float* __restrict__ bias, float* __restrict__ out,
    bf16* __restrict__ q_ws, bf16* __restrict__ k_ws, bf16* __restrict__ v_ws,
    int M, int N, int K) {
    __shared__ bf16 As[128][48];   // [m][k], 96 B row stride (16B aligned)
    __shared__ bf16 Bs[128][48];   // [n][k] (transposed weights)

    const int tid  = threadIdx.x;
    const int bm   = blockIdx.y * 128;
    const int bn   = blockIdx.x * 128;
    const int wid  = tid >> 5;
    const int lane = tid & 31;
    const int half = lane >> 4;
    const int lm   = lane & 15;
    const int wm   = (wid >> 2) * 64;
    const int wn   = (wid & 3) * 32;

    const v8f zero8 = {0.f, 0.f, 0.f, 0.f, 0.f, 0.f, 0.f, 0.f};
    v8f acc[4][2];
#pragma unroll
    for (int mt = 0; mt < 4; ++mt)
#pragma unroll
        for (int nt = 0; nt < 2; ++nt) acc[mt][nt] = zero8;

    for (int k0 = 0; k0 < K; k0 += 32) {
        // ---- stage A tile (128 x 32) ----
        if constexpr (sizeof(AT) == 4) {  // fp32 source: convert on stage
#pragma unroll
            for (int i = 0; i < 4; ++i) {
                int c = tid + i * 256;
                int row = c >> 3;
                int col = (c & 7) * 4;
                v4f av = *(const v4f*)(A + (size_t)(bm + row) * K + k0 + col);
                As[row][col + 0] = (bf16)av.x;
                As[row][col + 1] = (bf16)av.y;
                As[row][col + 2] = (bf16)av.z;
                As[row][col + 3] = (bf16)av.w;
            }
        } else {  // bf16 source: async DMA straight into LDS (ASYNCcnt)
#pragma unroll
            for (int i = 0; i < 2; ++i) {
                int c = tid + i * 256;
                int row = c >> 2;
                int col = (c & 3) * 8;
                async_load_b128(&As[row][col],
                                A + (size_t)(bm + row) * K + k0 + col);
            }
        }
        // ---- stage B tile transposed: Bs[n][k] <- B[k0+k][bn+n] ----
#pragma unroll
        for (int i = 0; i < 4; ++i) {
            int c = tid + i * 256;
            int k = c >> 5;
            int n4 = (c & 31) * 4;
            v4f bv = *(const v4f*)(Bw + (size_t)(k0 + k) * N + bn + n4);
            Bs[n4 + 0][k] = (bf16)bv.x;
            Bs[n4 + 1][k] = (bf16)bv.y;
            Bs[n4 + 2][k] = (bf16)bv.z;
            Bs[n4 + 3][k] = (bf16)bv.w;
        }
        if constexpr (sizeof(AT) == 2) wait_async0();
        __syncthreads();

        // ---- fragments + WMMA ----
        v16bf afrag[4], bfrag[2];
#pragma unroll
        for (int mt = 0; mt < 4; ++mt) {
            int row = wm + mt * 16 + lm;
            v8bf lo = *(const v8bf*)&As[row][half * 8];
            v8bf hi = *(const v8bf*)&As[row][half * 8 + 16];
            afrag[mt] = combine16(lo, hi);
        }
#pragma unroll
        for (int nt = 0; nt < 2; ++nt) {
            int col = wn + nt * 16 + lm;
            v8bf lo = *(const v8bf*)&Bs[col][half * 8];
            v8bf hi = *(const v8bf*)&Bs[col][half * 8 + 16];
            bfrag[nt] = combine16(lo, hi);
        }
#pragma unroll
        for (int mt = 0; mt < 4; ++mt)
#pragma unroll
            for (int nt = 0; nt < 2; ++nt)
                acc[mt][nt] = __builtin_amdgcn_wmma_f32_16x16x32_bf16(
                    false, afrag[mt], false, bfrag[nt], (short)0, acc[mt][nt],
                    false, false);
        __syncthreads();
    }

    // ---- epilogue ----
    // MODE 1: uniform per block which tensor (q/k/v) these columns feed.
    const int  t          = bn >> 10;                        // scalar (blockIdx)
    bf16*      dst        = (t == 0) ? q_ws : (t == 1) ? k_ws : v_ws;
    const float sc        = (t == 0) ? 0.125f : 1.0f;        // 1/sqrt(64) into Q
    const bool transposed = (t == 2);                        // V stored [B,H,D,S]

#pragma unroll
    for (int mt = 0; mt < 4; ++mt) {
#pragma unroll
        for (int nt = 0; nt < 2; ++nt) {
#pragma unroll
            for (int r = 0; r < 8; ++r) {
                int m = bm + wm + mt * 16 + r + half * 8;
                int n = bn + wn + nt * 16 + lm;
                float v = acc[mt][nt][r] + bias[n];
                if constexpr (MODE == 0) {
                    out[(size_t)m * N + n] = v;
                } else {
                    int bi = m >> 11;                 // / SEQ
                    int s  = m & (SEQ - 1);
                    int hh = (n >> 6) & (HEADS - 1);
                    int d  = n & (HDIM - 1);
                    size_t head  = (size_t)bi * HEADS + hh;
                    u32    inner = transposed ? (u32)(d * SEQ + s)
                                              : (u32)(s * HDIM + d);
                    dst[head * (SEQ * HDIM) + inner] = (bf16)(v * sc);
                }
            }
        }
    }
}

// ---------------------------------------------------------------------------
// Flash attention: WG per (b, h, 128-row q tile); 8 waves, 16 q rows each.
// K and V^T blocks staged by the Tensor Data Mover with double buffering:
// wave 0 issues next block's DMA, all waves compute on the current buffer.
// ---------------------------------------------------------------------------
__global__ __launch_bounds__(256) void attn_flash_wmma(
    const bf16* __restrict__ q_ws, const bf16* __restrict__ k_ws,
    const bf16* __restrict__ v_ws, bf16* __restrict__ attn_ws) {
    __shared__ bf16 Ks[2][64][80];  // [buf][key][d], 160 B rows (TDM-padded)
    __shared__ bf16 Vs[2][64][80];  // [buf][d][key], 160 B rows (TDM-padded)
    __shared__ bf16 Ps[8][16][80];  // per-wave P patch (C-layout -> A-layout)

    const int b  = blockIdx.z;
    const int h  = blockIdx.y;
    const int qt = blockIdx.x;
    const int tid  = threadIdx.x;
    const int wid  = tid >> 5;
    const int lane = tid & 31;
    const int half = lane >> 4;
    const int lm   = lane & 15;

    const size_t head_off = ((size_t)b * HEADS + h) * SEQ * HDIM;
    const bf16* Q  = q_ws + head_off + (size_t)(qt * 128) * HDIM;
    const bf16* Kp = k_ws + head_off;   // [S][D]
    const bf16* Vt = v_ws + head_off;   // [D][S] (pre-transposed by QKV epilogue)

    v16bf qfrag[2];
    {
        int row = wid * 16 + lm;
#pragma unroll
        for (int ks = 0; ks < 2; ++ks) {
            v8bf lo = *(const v8bf*)(Q + (size_t)row * HDIM + ks * 32 + half * 8);
            v8bf hi = *(const v8bf*)(Q + (size_t)row * HDIM + ks * 32 + half * 8 + 16);
            qfrag[ks] = combine16(lo, hi);
        }
    }

    const v8f zero8 = {0.f, 0.f, 0.f, 0.f, 0.f, 0.f, 0.f, 0.f};
    v8f o[4];
#pragma unroll
    for (int dt = 0; dt < 4; ++dt) o[dt] = zero8;
    float m_i[8], l_i[8];
#pragma unroll
    for (int r = 0; r < 8; ++r) { m_i[r] = -1e30f; l_i[r] = 0.f; }

    const u32 ks_lds[2] = {lds_addr32(&Ks[0][0][0]), lds_addr32(&Ks[1][0][0])};
    const u32 vs_lds[2] = {lds_addr32(&Vs[0][0][0]), lds_addr32(&Vs[1][0][0])};

    // Prologue: DMA block 0 into buffer 0.
    if (wid == 0) {
        tdm_load_tile64(ks_lds[0], Kp, HDIM);
        tdm_load_tile64(vs_lds[0], Vt, SEQ);
        wait_tensor0();
    }
    __syncthreads();

    for (int kb = 0; kb < SEQ; kb += 64) {
        const int buf = (kb >> 6) & 1;
        // Prefetch next block into the other buffer (overlaps with compute).
        if (wid == 0 && kb + 64 < SEQ) {
            tdm_load_tile64(ks_lds[buf ^ 1], Kp + (size_t)(kb + 64) * HDIM, HDIM);
            tdm_load_tile64(vs_lds[buf ^ 1], Vt + (kb + 64), SEQ);
        }

        // ---- scores S = Q * K^T (Q pre-scaled by 1/sqrt(D)) ----
        v8f sacc[4];
#pragma unroll
        for (int nt = 0; nt < 4; ++nt) sacc[nt] = zero8;
#pragma unroll
        for (int ks = 0; ks < 2; ++ks) {
#pragma unroll
            for (int nt = 0; nt < 4; ++nt) {
                int keyc = nt * 16 + lm;
                v8bf lo = *(const v8bf*)&Ks[buf][keyc][ks * 32 + half * 8];
                v8bf hi = *(const v8bf*)&Ks[buf][keyc][ks * 32 + half * 8 + 16];
                v16bf bf = combine16(lo, hi);
                sacc[nt] = __builtin_amdgcn_wmma_f32_16x16x32_bf16(
                    false, qfrag[ks], false, bf, (short)0, sacc[nt], false, false);
            }
        }

        // ---- online softmax (row = r + 8*half, col = lm + nt*16) ----
#pragma unroll
        for (int r = 0; r < 8; ++r) {
            float mx = sacc[0][r];
#pragma unroll
            for (int nt = 1; nt < 4; ++nt) mx = fmaxf(mx, sacc[nt][r]);
            mx = fmaxf(mx, swz_xor<1>(mx));
            mx = fmaxf(mx, swz_xor<2>(mx));
            mx = fmaxf(mx, swz_xor<4>(mx));
            mx = fmaxf(mx, swz_xor<8>(mx));
            float mnew = fmaxf(m_i[r], mx);
            float corr = __expf(m_i[r] - mnew);
            m_i[r] = mnew;
            float sum = 0.f;
#pragma unroll
            for (int nt = 0; nt < 4; ++nt) {
                float p = __expf(sacc[nt][r] - mnew);
                sum += p;
                Ps[wid][r + half * 8][nt * 16 + lm] = (bf16)p;
            }
            sum += swz_xor<1>(sum);
            sum += swz_xor<2>(sum);
            sum += swz_xor<4>(sum);
            sum += swz_xor<8>(sum);
            l_i[r] = l_i[r] * corr + sum;
#pragma unroll
            for (int dt = 0; dt < 4; ++dt) o[dt][r] *= corr;
        }
        __syncthreads();  // publish P patches (cross-lane via LDS)

        // ---- O += P * V ----
#pragma unroll
        for (int ks = 0; ks < 2; ++ks) {
            v8bf plo = *(const v8bf*)&Ps[wid][lm][ks * 32 + half * 8];
            v8bf phi = *(const v8bf*)&Ps[wid][lm][ks * 32 + half * 8 + 16];
            v16bf pf = combine16(plo, phi);
#pragma unroll
            for (int dt = 0; dt < 4; ++dt) {
                int dc = dt * 16 + lm;
                v8bf lo = *(const v8bf*)&Vs[buf][dc][ks * 32 + half * 8];
                v8bf hi = *(const v8bf*)&Vs[buf][dc][ks * 32 + half * 8 + 16];
                v16bf vf = combine16(lo, hi);
                o[dt] = __builtin_amdgcn_wmma_f32_16x16x32_bf16(
                    false, pf, false, vf, (short)0, o[dt], false, false);
            }
        }
        // Wait for the prefetch DMA, then let everyone flip buffers.
        if (wid == 0) wait_tensor0();
        __syncthreads();
    }

    // ---- finalize: attn[b][s][h*64+d] bf16 for the output GEMM ----
    bf16* outp = attn_ws + ((size_t)b * SEQ + qt * 128 + wid * 16) * EMBED + h * HDIM;
#pragma unroll
    for (int dt = 0; dt < 4; ++dt) {
#pragma unroll
        for (int r = 0; r < 8; ++r) {
            int row = r + half * 8;
            int d = dt * 16 + lm;
            outp[(size_t)row * EMBED + d] = (bf16)(o[dt][r] / l_i[r]);
        }
    }
}

// ---------------------------------------------------------------------------
// Launch: QKV GEMM -> flash attention -> output GEMM
// Workspace (bf16): q | k | v (each B*H*S*D) | attn (B*S*E)  = 32 MB
// ---------------------------------------------------------------------------
extern "C" void kernel_launch(void* const* d_in, const int* in_sizes, int n_in,
                              void* d_out, int out_size, void* d_ws, size_t ws_size,
                              hipStream_t stream) {
    const float* x     = (const float*)d_in[0];
    const float* w_qkv = (const float*)d_in[1];
    const float* b_qkv = (const float*)d_in[2];
    const float* w_out = (const float*)d_in[3];
    const float* b_out = (const float*)d_in[4];
    float* out = (float*)d_out;

    const size_t per_tensor = (size_t)BATCH * HEADS * SEQ * HDIM;
    bf16* q_ws    = (bf16*)d_ws;
    bf16* k_ws    = q_ws + per_tensor;
    bf16* v_ws    = k_ws + per_tensor;
    bf16* attn_ws = v_ws + per_tensor;

    const int M = BATCH * SEQ;  // 4096

    dim3 g1(3 * EMBED / 128, M / 128);
    gemm_bf16_wmma<float, 1><<<g1, 256, 0, stream>>>(
        x, w_qkv, b_qkv, nullptr, q_ws, k_ws, v_ws, M, 3 * EMBED, EMBED);

    dim3 g2(SEQ / 128, HEADS, BATCH);
    attn_flash_wmma<<<g2, 256, 0, stream>>>(q_ws, k_ws, v_ws, attn_ws);

    dim3 g3(EMBED / 128, M / 128);
    gemm_bf16_wmma<bf16, 0><<<g3, 256, 0, stream>>>(
        attn_ws, w_out, b_out, out, nullptr, nullptr, nullptr, M, EMBED, EMBED);
}
